// Mnist_Classifier_27960237097613
// MI455X (gfx1250) — matrix operand order repaired and checked by reading, not compiled, not run on hardware
//
#include <hip/hip_runtime.h>
#include <math.h>

typedef _Float16 h16;
typedef __attribute__((ext_vector_type(16))) _Float16 v16h;
typedef __attribute__((ext_vector_type(8)))  _Float16 v8h;
typedef __attribute__((ext_vector_type(8)))  float    v8f;

#define PI_D 3.14159265358979323846264338327950288
#define PI_F 3.14159265358979323846f

// ---------------- Wigner small-d (double precision, explicit sum) -----------
__device__ __forceinline__ double factd(int n) {
    double r = 1.0;
    for (int i = 2; i <= n; ++i) r *= (double)i;
    return r;
}
__device__ __forceinline__ double dpowi(double x, int n) {
    double r = 1.0;
    for (int i = 0; i < n; ++i) r *= x;
    return r;
}
// d^l_{mp,m}(beta) = <l mp| e^{-i beta Jy} |l m>, standard convention
__device__ double wigner_d(int l, int mp, int m, double beta) {
    if (mp < -l || mp > l || m < -l || m > l) return 0.0;
    double cb = cos(0.5 * beta), sb = sin(0.5 * beta);
    double pref = sqrt(factd(l + mp) * factd(l - mp) * factd(l + m) * factd(l - m));
    int smin = (m - mp > 0) ? (m - mp) : 0;
    int smax = (l + m < l - mp) ? (l + m) : (l - mp);
    double sum = 0.0;
    for (int s = smin; s <= smax; ++s) {
        double den = factd(l + m - s) * factd(s) * factd(mp - m + s) * factd(l - mp - s);
        double t = dpowi(cb, 2 * l + m - mp - 2 * s) * dpowi(sb, mp - m + 2 * s) / den;
        sum += ((mp - m + s) & 1) ? -t : t;
    }
    return pref * sum;
}

// ---------------- Setup kernels ---------------------------------------------
__global__ void k_quad(float* w30, float* wq20, float* wq10) {
    int t = blockIdx.x * blockDim.x + threadIdx.x;
    int b, k; float* dst; int di;
    if      (t < 60) { b = 30; k = t;      dst = w30;  di = t; }
    else if (t < 80) { b = 10; k = t - 60; dst = wq20; di = t - 60; }
    else if (t < 90) { b = 5;  k = t - 80; dst = wq10; di = t - 80; }
    else return;
    double beta = PI_D * (2 * k + 1) / (4.0 * b);
    double s = 0.0;
    for (int j = 0; j < b; ++j) s += sin((2 * j + 1) * beta) / (2 * j + 1);
    dst[di] = (float)(2.0 / b * sin(beta) * s);
}

// dS2w[l][k][mi] = d^l_{m,0}(beta30_k) * w30[k], mi = m+9, [10][60][19]
__global__ void k_ds2w(const float* __restrict__ w30, float* __restrict__ dS2w) {
    int t = blockIdx.x * blockDim.x + threadIdx.x;
    if (t >= 10 * 60 * 19) return;
    int mi = t % 19, k = (t / 19) % 60, l = t / (19 * 60);
    double beta = PI_D * (2 * k + 1) / 120.0;
    dS2w[t] = (float)wigner_d(l, mi - 9, 0, beta) * w30[k];
}

// dL1[l][k][mi][ni] = d^l_{m,n}(beta10_k), [10][20][19][19]
__global__ void k_dL1(float* __restrict__ dL1) {
    int t = blockIdx.x * blockDim.x + threadIdx.x;
    if (t >= 10 * 20 * 19 * 19) return;
    int ni = t % 19, mi = (t / 19) % 19, k = (t / 361) % 20, l = t / (361 * 20);
    double beta = PI_D * (2 * k + 1) / 40.0;
    dL1[t] = (float)wigner_d(l, mi - 9, ni - 9, beta);
}

// dL2[l][k][mi][ni] = d^l_{m,n}(beta5_k), [5][10][9][9]
__global__ void k_dL2(float* __restrict__ dL2) {
    int t = blockIdx.x * blockDim.x + threadIdx.x;
    if (t >= 5 * 10 * 9 * 9) return;
    int ni = t % 9, mi = (t / 9) % 9, k = (t / 81) % 10, l = t / 810;
    double beta = PI_D * (2 * k + 1) / 20.0;
    dL2[t] = (float)wigner_d(l, mi - 4, ni - 4, beta);
}

// ---------------- Stage 1: S2 FFT -------------------------------------------
__global__ void k_fsel(const float* __restrict__ x, float* __restrict__ Fsel) {
    int t = blockIdx.x * blockDim.x + threadIdx.x;
    if (t >= 16 * 60 * 19) return;
    int mi = t % 19, k = (t / 19) % 60, z = t / (19 * 60);
    int m = mi - 9;
    const float* xp = x + ((size_t)z * 60 + k) * 60;
    float re = 0.f, im = 0.f;
    for (int g = 0; g < 60; ++g) {
        int mg = ((m * g) % 60 + 60) % 60;
        float th = -(2.0f * PI_F / 60.0f) * (float)mg;
        float v = xp[g];
        re += v * cosf(th);
        im += v * sinf(th);
    }
    Fsel[2 * t] = re; Fsel[2 * t + 1] = im;
}

__global__ void k_xh(const float* __restrict__ dS2w, const float* __restrict__ Fsel,
                     float* __restrict__ xh) {
    int t = blockIdx.x * blockDim.x + threadIdx.x;
    if (t >= 10 * 16 * 19) return;
    int mi = t % 19, z = (t / 19) % 16, l = t / (19 * 16);
    float sre = 0.f, sim = 0.f;
    for (int k = 0; k < 60; ++k) {
        float d = dS2w[(l * 60 + k) * 19 + mi];   // zero for |m|>l
        sre += d * Fsel[(((size_t)z * 60 + k) * 19 + mi) * 2];
        sim += d * Fsel[(((size_t)z * 60 + k) * 19 + mi) * 2 + 1];
    }
    xh[2 * t] = sre; xh[2 * t + 1] = sim;
}

// ---------------- Stage 3 GEMM operands -------------------------------------
__global__ void k_A1(const float* __restrict__ dL1, const float* __restrict__ xh,
                     h16* __restrict__ A1) {
    int t = blockIdx.x * blockDim.x + threadIdx.x;
    if (t >= 6400 * 100) return;
    int c = t % 100, row = t / 100;
    int z = row / 400, ka = row % 400, k = ka / 20, a = ka % 20;
    int l = 0; while ((l + 1) * (l + 1) <= c) ++l;
    int n = c - l * l - l;
    float Rre = 0.f, Rim = 0.f;
    for (int m = -l; m <= l; ++m) {
        float d = dL1[((l * 20 + k) * 19 + (m + 9)) * 19 + (n + 9)];
        float xr = xh[((l * 16 + z) * 19 + (m + 9)) * 2];
        float xi = xh[((l * 16 + z) * 19 + (m + 9)) * 2 + 1];
        int ph = ((m * a) % 20 + 20) % 20;
        float th = (2.0f * PI_F / 20.0f) * (float)ph;
        float cr = cosf(th), sr = sinf(th);
        Rre += d * (xr * cr - xi * sr);
        Rim += d * (xr * sr + xi * cr);
    }
    A1[(size_t)row * 200 + 2 * c]     = (h16)Rre;
    A1[(size_t)row * 200 + 2 * c + 1] = (h16)Rim;
}

__global__ void k_B1(const float* __restrict__ p_re, const float* __restrict__ p_im,
                     h16* __restrict__ B1) {
    int t = blockIdx.x * blockDim.x + threadIdx.x;
    if (t >= 200 * 2000) return;
    int col = t % 2000, rowr = t / 2000;
    int c = rowr >> 1, ri = rowr & 1;
    int l = 0; while ((l + 1) * (l + 1) <= c) ++l;
    int n = c - l * l - l;
    int o = col / 20, g = col % 20;
    int ph = ((n * g) % 20 + 20) % 20;
    float th = (2.0f * PI_F / 20.0f) * (float)ph;
    float pr = p_re[o * 100 + l * l + n + l];
    float pi = p_im[o * 100 + l * l + n + l];
    float coef = (float)(2 * l + 1);
    float ct = cosf(th), st = sinf(th);
    float v = (ri == 0) ? coef * (pr * ct + pi * st) : coef * (pi * ct - pr * st);
    B1[(size_t)rowr * 2000 + col] = (h16)v;
}

// ---------------- generic f16 WMMA GEMM -------------------------------------
// C = A(MxK, row stride lda) @ B(KxN, row stride ldb).
// mode 0: f32 store to Cf. mode 1 (GEMM1 only): relu + f16 permuted store:
//   row=z*400+k*20+a, col=o*20+g -> h1p[(z*100+o)*8000 + (k*20+a)*20+g]
// Block: 256 thr = 8 waves (2 M x 4 N), tile 32(M) x 128(N), K-stage 64.
// B is staged K-major (transposed) in LDS so all fragment reads are b128.
__global__ __launch_bounds__(256) void k_gemm(const h16* __restrict__ A, int lda,
                                              const h16* __restrict__ B, int ldb,
                                              float* __restrict__ Cf,
                                              h16* __restrict__ Ch,
                                              int M, int N, int K, int mode) {
    __shared__ __align__(16) h16 As[32 * 72];    // 32 M-rows x 64 K (+8 pad)
    __shared__ __align__(16) h16 Bst[128 * 72];  // 128 N-rows x 64 K (+8 pad)
    int tid = threadIdx.x;
    int wave = tid >> 5, lane = tid & 31;
    int wm = wave >> 2, wn = wave & 3;
    int half = lane >> 4, l15 = lane & 15;
    int bm = blockIdx.x * 32, bn = blockIdx.y * 128;
    v8f acc0 = {0.f, 0.f, 0.f, 0.f, 0.f, 0.f, 0.f, 0.f};
    v8f acc1 = {0.f, 0.f, 0.f, 0.f, 0.f, 0.f, 0.f, 0.f};
    for (int k0 = 0; k0 < K; k0 += 64) {
        bool full = (bm + 32 <= M) && (bn + 128 <= N) && (k0 + 64 <= K);
        if (full) {
            // A: 32x64 halves = 256 x v8h, one b128 load+store per thread
            {
                int r = tid >> 3, c8 = (tid & 7) << 3;
                *(v8h*)&As[r * 72 + c8] =
                    *(const v8h*)&A[(size_t)(bm + r) * lda + k0 + c8];
            }
            // B: 64(K) x 128(N): each thread four v8h loads along N, scatter K-major
            {
                int c8 = (tid & 15) << 3;
                #pragma unroll
                for (int rr = 0; rr < 4; ++rr) {
                    int kr = (tid >> 4) + rr * 16;
                    v8h v = *(const v8h*)&B[(size_t)(k0 + kr) * ldb + bn + c8];
                    #pragma unroll
                    for (int e = 0; e < 8; ++e) Bst[(c8 + e) * 72 + kr] = v[e];
                }
            }
        } else {
            for (int tt = tid; tt < 32 * 64; tt += 256) {
                int r = tt >> 6, c = tt & 63;
                int gr = bm + r, gc = k0 + c;
                As[r * 72 + c] = (gr < M && gc < K) ? A[(size_t)gr * lda + gc] : (h16)0.0f;
            }
            for (int tt = tid; tt < 64 * 128; tt += 256) {
                int kr = tt >> 7, c = tt & 127;
                int gk = k0 + kr, gn = bn + c;
                Bst[c * 72 + kr] = (gk < K && gn < N) ? B[(size_t)gk * ldb + gn] : (h16)0.0f;
            }
        }
        __syncthreads();
        const h16* Ab  = &As[(wm * 16 + l15) * 72];
        const h16* Bb0 = &Bst[(wn * 32 + l15) * 72];
        const h16* Bb1 = &Bst[(wn * 32 + 16 + l15) * 72];
        #pragma unroll
        for (int ks = 0; ks < 2; ++ks) {
            // A frag: lane(M=l15+?, half) needs K = {16h..16h+15}, contiguous
            v8h a0 = *(const v8h*)&Ab[ks * 32 + 16 * half];
            v8h a1 = *(const v8h*)&Ab[ks * 32 + 16 * half + 8];
            v16h af = __builtin_shufflevector(a0, a1, 0,1,2,3,4,5,6,7,8,9,10,11,12,13,14,15);
            // B frag: lane(N=l15, half) needs K = {16h..16h+15}, contiguous
            v8h p0 = *(const v8h*)&Bb0[ks * 32 + 16 * half];
            v8h p1 = *(const v8h*)&Bb0[ks * 32 + 16 * half + 8];
            v16h bf0 = __builtin_shufflevector(p0, p1, 0,1,2,3,4,5,6,7,8,9,10,11,12,13,14,15);
            v8h q0 = *(const v8h*)&Bb1[ks * 32 + 16 * half];
            v8h q1 = *(const v8h*)&Bb1[ks * 32 + 16 * half + 8];
            v16h bf1 = __builtin_shufflevector(q0, q1, 0,1,2,3,4,5,6,7,8,9,10,11,12,13,14,15);
            acc0 = __builtin_amdgcn_wmma_f32_16x16x32_f16(false, af, false, bf0, (short)0, acc0, false, false);
            acc1 = __builtin_amdgcn_wmma_f32_16x16x32_f16(false, af, false, bf1, (short)0, acc1, false, false);
        }
        __syncthreads();
    }
    // D layout: VGPR r -> M = r + 8*half, N = lane&15
    #pragma unroll
    for (int r = 0; r < 8; ++r) {
        int dm = bm + wm * 16 + r + 8 * half;
        #pragma unroll
        for (int s = 0; s < 2; ++s) {
            int dn = bn + wn * 32 + s * 16 + l15;
            if (dm < M && dn < N) {
                float v = s ? acc1[r] : acc0[r];
                if (mode == 0) {
                    Cf[(size_t)dm * N + dn] = v;
                } else {
                    v = v > 0.f ? v : 0.f;                 // relu(h1)
                    int z = dm / 400, ka = dm % 400;       // ka = k*20+a
                    int o = dn / 20,  g  = dn % 20;
                    Ch[(size_t)(z * 100 + o) * 8000 + ka * 20 + g] = (h16)v;
                }
            }
        }
    }
}

// ---------------- Stage 4 weight matrix (row stride 336 for alignment) -------
__global__ void k_B2(const float* __restrict__ wq20, const float* __restrict__ dL1,
                     h16* __restrict__ B2) {
    int t = blockIdx.x * blockDim.x + threadIdx.x;
    if (t >= 8000 * 330) return;
    int col = t % 330, row = t / 330;
    int c = col >> 1, ri = col & 1;
    const int off[6] = {0, 1, 10, 35, 84, 165};
    int l = 0; while (l < 4 && c >= off[l + 1]) ++l;
    int dim = 2 * l + 1, r = c - off[l];
    int m = r / dim - l, n = r % dim - l;
    int k = row / 400, ag = row % 400, a = ag / 20, g = ag % 20;
    int ph = ((m * a + n * g) % 20 + 20) % 20;
    float th = (2.0f * PI_F / 20.0f) * (float)ph;
    float coef = wq20[k] * dL1[((l * 20 + k) * 19 + m + 9) * 19 + n + 9] * (1.0f / 400.0f);
    float v = ri ? -coef * sinf(th) : coef * cosf(th);
    B2[(size_t)row * 336 + col] = (h16)v;
}

// ---------------- Stage 5 operands (per l) ----------------------------------
__global__ void k_A5(const float* __restrict__ GH, h16* __restrict__ A5,
                     int dim, int K5, int off5l, int total) {
    int t = blockIdx.x * blockDim.x + threadIdx.x;
    if (t >= total) return;
    int k5 = t % K5, row = t / K5;
    int z = row / dim, mi = row % dim;
    int part = k5 / (100 * dim), rem = k5 % (100 * dim);
    int i = rem / dim, ki = rem % dim;
    float v = GH[(size_t)(z * 100 + i) * 330 + 2 * (off5l + mi * dim + ki) + part];
    A5[(size_t)row * K5 + k5] = (h16)v;
}

__global__ void k_B5(const float* __restrict__ p_re, const float* __restrict__ p_im,
                     h16* __restrict__ B5, int dim, int K5, int N5, int off5l, int total) {
    int t = blockIdx.x * blockDim.x + threadIdx.x;
    if (t >= total) return;
    int col = t % N5, k5 = t / N5;
    int part = k5 / (100 * dim), rem = k5 % (100 * dim);
    int i = rem / dim, ki = rem % dim;
    int o = col / (2 * dim), rem2 = col % (2 * dim);
    int ni = rem2 >> 1, ri = rem2 & 1;
    size_t p = (size_t)(o * 100 + i) * 165 + off5l + ni * dim + ki;
    float pr = p_re[p], pi = p_im[p];
    float v = (ri == 0) ? (part == 0 ? pr : pi) : (part == 0 ? -pi : pr);
    B5[(size_t)k5 * N5 + col] = (h16)v;
}

// A6 row stride 336 for alignment
__global__ void k_A6(const float* __restrict__ Z2, h16* __restrict__ A6,
                     int dim, int N5, int off5l, int total) {
    int t = blockIdx.x * blockDim.x + threadIdx.x;
    if (t >= total) return;
    int ri = t & 1; int tmp = t >> 1;
    int ni = tmp % dim; tmp /= dim;
    int mi = tmp % dim; tmp /= dim;
    int o = tmp % 200;  int z = tmp / 200;
    float v = Z2[(size_t)(z * dim + mi) * N5 + (o * dim + ni) * 2 + ri];
    A6[(size_t)(z * 200 + o) * 336 + 2 * (off5l + mi * dim + ni) + ri] = (h16)v;
}

__global__ void k_B6(const float* __restrict__ dL2, h16* __restrict__ B6) {
    int t = blockIdx.x * blockDim.x + threadIdx.x;
    if (t >= 330 * 1000) return;
    int col = t % 1000, rowr = t / 1000;
    int c = rowr >> 1, ri = rowr & 1;
    const int off[6] = {0, 1, 10, 35, 84, 165};
    int l = 0; while (l < 4 && c >= off[l + 1]) ++l;
    int dim = 2 * l + 1, r = c - off[l];
    int m = r / dim - l, n = r % dim - l;
    int k = col / 100, ag = col % 100, a = ag / 10, g = ag % 10;
    int ph = ((m * a + n * g) % 10 + 10) % 10;
    float th = (2.0f * PI_F / 10.0f) * (float)ph;
    float coef = (float)(2 * l + 1) * dL2[((l * 10 + k) * 9 + m + 4) * 9 + n + 4];
    float v = ri ? -coef * sinf(th) : coef * cosf(th);
    B6[(size_t)rowr * 1000 + col] = (h16)v;
}

__global__ void k_feat(const float* __restrict__ H2, const float* __restrict__ wq10,
                       float* __restrict__ feat) {
    int t = blockIdx.x * blockDim.x + threadIdx.x;
    if (t >= 3200) return;
    const float* hp = H2 + (size_t)t * 1000;
    float s = 0.f;
    for (int k = 0; k < 10; ++k) {
        float ss = 0.f;
        for (int q = 0; q < 100; ++q) {
            float v = hp[k * 100 + q];
            ss += v > 0.f ? v : 0.f;
        }
        s += wq10[k] * 0.01f * ss;
    }
    feat[t] = s;
}

__global__ void k_out(const float* __restrict__ feat, const float* __restrict__ W,
                      const float* __restrict__ bias, float* __restrict__ out) {
    int t = blockIdx.x * blockDim.x + threadIdx.x;
    if (t >= 160) return;
    int z = t / 10, j = t % 10;
    float s = bias[j];
    for (int o = 0; o < 200; ++o) s += feat[z * 200 + o] * W[j * 200 + o];
    out[t] = s;
}

// ---------------------------------------------------------------------------
extern "C" void kernel_launch(void* const* d_in, const int* in_sizes, int n_in,
                              void* d_out, int out_size, void* d_ws, size_t ws_size,
                              hipStream_t stream) {
    const float* x        = (const float*)d_in[0];
    const float* psi1_re  = (const float*)d_in[1];
    const float* psi1_im  = (const float*)d_in[2];
    const float* psi2_re  = (const float*)d_in[3];
    const float* psi2_im  = (const float*)d_in[4];
    const float* Wlin     = (const float*)d_in[5];
    const float* bias     = (const float*)d_in[6];
    float* out = (float*)d_out;

    char* base = (char*)d_ws;
    size_t off = 0;
    auto alloc = [&](size_t bytes) -> char* {
        off = (off + 255) & ~(size_t)255;
        char* r = base + off;
        off += bytes;
        return r;
    };
    float* w30  = (float*)alloc(60 * 4);
    float* wq20 = (float*)alloc(20 * 4);
    float* wq10 = (float*)alloc(10 * 4);
    float* dS2w = (float*)alloc((size_t)10 * 60 * 19 * 4);
    float* dL1  = (float*)alloc((size_t)10 * 20 * 19 * 19 * 4);
    float* dL2  = (float*)alloc((size_t)5 * 10 * 9 * 9 * 4);
    float* Fsel = (float*)alloc((size_t)16 * 60 * 19 * 2 * 4);
    float* xh   = (float*)alloc((size_t)10 * 16 * 19 * 2 * 4);
    h16*   A1   = (h16*)alloc((size_t)6400 * 200 * 2);
    h16*   B1   = (h16*)alloc((size_t)200 * 2000 * 2);
    h16*   h1p  = (h16*)alloc((size_t)1600 * 8000 * 2);
    h16*   B2   = (h16*)alloc((size_t)8000 * 336 * 2);   // ldb = 336
    float* GH   = (float*)alloc((size_t)1600 * 330 * 4);
    h16*   A5   = (h16*)alloc((size_t)144 * 1800 * 2);
    h16*   B5   = (h16*)alloc((size_t)1800 * 3600 * 2);
    float* Z2   = (float*)alloc((size_t)144 * 3600 * 4);
    h16*   A6   = (h16*)alloc((size_t)3200 * 336 * 2);   // lda = 336
    h16*   B6   = (h16*)alloc((size_t)330 * 1000 * 2);
    float* H2   = (float*)alloc((size_t)3200 * 1000 * 4);
    float* feat = (float*)alloc((size_t)3200 * 4);
    (void)ws_size; (void)in_sizes; (void)n_in; (void)out_size;

    auto blks = [](int n) { return (n + 255) / 256; };

    // Setup: quadrature + Wigner matrices
    k_quad<<<1, 128, 0, stream>>>(w30, wq20, wq10);
    k_ds2w<<<blks(10 * 60 * 19), 256, 0, stream>>>(w30, dS2w);
    k_dL1 <<<blks(10 * 20 * 19 * 19), 256, 0, stream>>>(dL1);
    k_dL2 <<<blks(5 * 10 * 9 * 9), 256, 0, stream>>>(dL2);

    // Stage 1: S2 FFT + projection
    k_fsel<<<blks(16 * 60 * 19), 256, 0, stream>>>(x, Fsel);
    k_xh  <<<blks(10 * 16 * 19), 256, 0, stream>>>(dS2w, Fsel, xh);

    // Stage 3: GEMM1 (6400x200 @ 200x2000) -> relu+permute -> h1p (f16)
    k_A1<<<blks(6400 * 100), 256, 0, stream>>>(dL1, xh, A1);
    k_B1<<<blks(200 * 2000), 256, 0, stream>>>(psi1_re, psi1_im, B1);
    k_gemm<<<dim3(6400 / 32, (2000 + 127) / 128), 256, 0, stream>>>(
        A1, 200, B1, 2000, nullptr, h1p, 6400, 2000, 200, 1);

    // Stage 4: SO3 FFT as GEMM2 (1600x8000 @ 8000x330) -> GH (f32)
    k_B2<<<blks(8000 * 330), 256, 0, stream>>>(wq20, dL1, B2);
    k_gemm<<<dim3(1600 / 32, (330 + 127) / 128), 256, 0, stream>>>(
        h1p, 8000, B2, 336, GH, nullptr, 1600, 330, 8000, 0);

    // Stage 5: per-l complex contractions as real GEMMs, scatter into A6
    const int off5[6] = {0, 1, 10, 35, 84, 165};
    for (int l = 0; l < 5; ++l) {
        int dim = 2 * l + 1;
        int M5 = 16 * dim, K5 = 200 * dim, N5 = 400 * dim;
        int ta = M5 * K5, tb = K5 * N5, t6 = 3200 * dim * dim * 2;
        k_A5<<<blks(ta), 256, 0, stream>>>(GH, A5, dim, K5, off5[l], ta);
        k_B5<<<blks(tb), 256, 0, stream>>>(psi2_re, psi2_im, B5, dim, K5, N5, off5[l], tb);
        k_gemm<<<dim3((M5 + 31) / 32, (N5 + 127) / 128), 256, 0, stream>>>(
            A5, K5, B5, N5, Z2, nullptr, M5, N5, K5, 0);
        k_A6<<<blks(t6), 256, 0, stream>>>(Z2, A6, dim, N5, off5[l], t6);
    }

    // Stage 6: SO3 iFFT as GEMM3 (3200x330 @ 330x1000) -> H2 (f32)
    k_B6<<<blks(330 * 1000), 256, 0, stream>>>(dL2, B6);
    k_gemm<<<dim3(3200 / 32, (1000 + 127) / 128), 256, 0, stream>>>(
        A6, 336, B6, 1000, H2, nullptr, 3200, 1000, 330, 0);

    // Stage 7: relu + quadrature reduce, then linear head
    k_feat<<<blks(3200), 256, 0, stream>>>(H2, wq10, feat);
    k_out <<<1, 256, 0, stream>>>(feat, Wlin, bias, out);
}
